// DrugSynergyModel_43258910605928
// MI455X (gfx1250) — matrix operand order repaired and checked by reading, not compiled, not run on hardware
//
#include <hip/hip_runtime.h>
#include <hip/hip_bf16.h>

// DrugSynergyModel attention: B=8, S=1024, D=256, H=8, DEPTH=32.
// Outputs (concatenated in d_out): out [B,S,D] f32, attn [B,H,S,S] f32.
//
// MI455X roofline: 268 MB attn write dominates -> memory bound at 23.3 TB/s
// (~16 us floor). bf16 WMMA 16x16x32 (f32 accum) keeps the 13 GFLOP of
// matmul work far below the matrix ceiling. All WMMA fragments are loaded
// as 16-byte vectors (two b128 per fragment) by making every B operand
// K-contiguous (V transposed to [B,H,DEPTH,S]; weights transposed to
// [out][in] bf16).

#define Bn 8
#define Sn 1024
#define Dn 256
#define Hn 8
#define DEPTHn 32
#define LDST 1028  // LDS row stride (floats): 1028 % 64 = 4 -> rows m, m+8
                   // land on different banks during WMMA C writes

typedef __attribute__((ext_vector_type(8)))  __bf16 v8bf;
typedef __attribute__((ext_vector_type(16))) __bf16 v16bf;
typedef __attribute__((ext_vector_type(8)))  float  v8f;

__device__ __forceinline__ v16bf cat8(v8bf lo, v8bf hi) {
  return __builtin_shufflevector(lo, hi, 0, 1, 2, 3, 4, 5, 6, 7,
                                 8, 9, 10, 11, 12, 13, 14, 15);
}

// A fragment (16-bit 16x32, ISA 7.12.2): lane m=lane&15, half=lane>>4.
// Lane data = two contiguous 16B chunks at K = 8*half and K = 16+8*half.
__device__ __forceinline__ v16bf load_frag_a(const __bf16* base, int ld,
                                             int m, int half) {
  const __bf16* p = base + (size_t)m * ld + 8 * half;
  v8bf lo = *(const v8bf*)p;
  v8bf hi = *(const v8bf*)(p + 16);
  return cat8(lo, hi);
}

// B fragment (16-bit 32x16): same K order as A fragment. With K-contiguous
// storage (column stride ld) the lane's data is two contiguous 16B chunks
// at K = 8*half and K = 16+8*half.
__device__ __forceinline__ v16bf load_frag_b(const __bf16* base, int ld,
                                             int n, int half) {
  const __bf16* p = base + (size_t)n * ld + 8 * half;
  v8bf lo = *(const v8bf*)p;
  v8bf hi = *(const v8bf*)(p + 16);
  return cat8(lo, hi);
}

// A fragment from f32 row-major source (float4 loads + pack-convert).
__device__ __forceinline__ v16bf load_frag_a_f32(const float* base, int ld,
                                                 int m, int half) {
  const float* p = base + (size_t)m * ld + 8 * half;
  float4 f0 = ((const float4*)p)[0];
  float4 f1 = ((const float4*)p)[1];
  float4 f2 = ((const float4*)(p + 16))[0];
  float4 f3 = ((const float4*)(p + 16))[1];
  v16bf a;
  a[0] = (__bf16)f0.x;  a[1] = (__bf16)f0.y;  a[2] = (__bf16)f0.z;  a[3] = (__bf16)f0.w;
  a[4] = (__bf16)f1.x;  a[5] = (__bf16)f1.y;  a[6] = (__bf16)f1.z;  a[7] = (__bf16)f1.w;
  a[8] = (__bf16)f2.x;  a[9] = (__bf16)f2.y;  a[10] = (__bf16)f2.z; a[11] = (__bf16)f2.w;
  a[12] = (__bf16)f3.x; a[13] = (__bf16)f3.y; a[14] = (__bf16)f3.z; a[15] = (__bf16)f3.w;
  return a;
}

__device__ __forceinline__ v8f wmma_bf16(v16bf a, v16bf b, v8f c) {
  return __builtin_amdgcn_wmma_f32_16x16x32_bf16(
      false, a, false, b, (short)0, c, false, false);
}

// ---- Weight transpose + bf16 convert: w[in][out] f32 -> wt[out][in] bf16 --
__global__ void wtrans_kernel(const float* __restrict__ w,
                              __bf16* __restrict__ wt) {
  __shared__ float t[32][33];
  const int c = threadIdx.x & 31, r = threadIdx.x >> 5;  // 256 thr: 8 rows/pass
  const int k0 = blockIdx.x * 32, n0 = blockIdx.y * 32;
#pragma unroll
  for (int i = 0; i < 32; i += 8)
    t[r + i][c] = w[(size_t)(k0 + r + i) * Dn + n0 + c];
  __syncthreads();
#pragma unroll
  for (int i = 0; i < 32; i += 8)
    wt[(size_t)(n0 + r + i) * Dn + k0 + c] = (__bf16)t[c][r + i];
}

// ---- V transpose: [B,H,S,DEPTH] bf16 -> [B,H,DEPTH,S] bf16 ----------------
__global__ void vtrans_kernel(const __bf16* __restrict__ v,
                              __bf16* __restrict__ vt) {
  __shared__ __bf16 t[32][34];
  const int c = threadIdx.x & 31, r = threadIdx.x >> 5;
  const int bh = blockIdx.y, s0 = blockIdx.x * 32;
#pragma unroll
  for (int i = 0; i < 32; i += 8)
    t[r + i][c] = v[((size_t)bh * Sn + s0 + r + i) * DEPTHn + c];
  __syncthreads();
#pragma unroll
  for (int i = 0; i < 32; i += 8)
    vt[((size_t)bh * DEPTHn + r + i) * Sn + s0 + c] = t[c][r + i];
}

// ---- QKV projection: y = x @ W + b, stored bf16 [B,H,S,DEPTH] -------------
__global__ void proj_qkv_kernel(const float* __restrict__ x,
                                const __bf16* __restrict__ wt,  // [out][in]
                                const float* __restrict__ bias,
                                __bf16* __restrict__ dst) {
  const int lane = threadIdx.x & 31;
  const int m = lane & 15, half = lane >> 4;
  const int row0 = blockIdx.x * 16;  // rows of [B*S, D]
  const int col0 = blockIdx.y * 16;  // output columns

  v8f c = {};
  for (int kc = 0; kc < Dn; kc += 32) {
    v16bf a = load_frag_a_f32(x + (size_t)row0 * Dn + kc, Dn, m, half);
    v16bf b = load_frag_b(wt + (size_t)col0 * Dn + kc, Dn, m, half);
    c = wmma_bf16(a, b, c);
  }

  const int col = col0 + m;
  const float bv = bias[col];
  const int h = col >> 5, dep = col & 31;
#pragma unroll
  for (int j = 0; j < 8; ++j) {
    const int r = row0 + j + (half ? 8 : 0);
    const int bb = r >> 10, s = r & (Sn - 1);
    dst[((size_t)(bb * Hn + h) * Sn + s) * DEPTHn + dep] = (__bf16)(c[j] + bv);
  }
}

// ---- Attention: WMMA logits -> biases+softmax -> attn out -> P@V ----------
__global__ void attn_kernel(const __bf16* __restrict__ qbf,
                            const __bf16* __restrict__ kbf,
                            const __bf16* __restrict__ vt,  // [B,H,DEPTH,S]
                            const float* __restrict__ adj,
                            const float* __restrict__ dist,
                            float* __restrict__ attn_out,
                            __bf16* __restrict__ ctx) {
  extern __shared__ float smem[];
  const int lane = threadIdx.x & 31;
  const int wv = threadIdx.x >> 5;  // wave 0..3 -> 16 query rows each
  const int m = lane & 15, half = lane >> 4;
  const int bh = blockIdx.y, b = bh >> 3, h = bh & 7;
  const int q0 = blockIdx.x * 64 + wv * 16;

  float* L = smem + (size_t)wv * 16 * LDST;  // [16][LDST]

  const float scale = 0.17677669529663689f;  // 1/sqrt(32)
  const size_t qrow_base = (size_t)b * Sn + q0;

  // Q tile A fragment, resident for all 64 K-tiles
  v16bf aq = load_frag_a(qbf + ((size_t)bh * Sn + q0) * DEPTHn, DEPTHn, m, half);

  // Phase 1: logits = scale * Q K^T. B = K^T is depth(K)-contiguous per key.
  for (int kt = 0; kt < Sn / 16; ++kt) {
    v16bf bk = load_frag_b(kbf + ((size_t)bh * Sn + kt * 16) * DEPTHn,
                           DEPTHn, m, half);
    v8f c = {};
    c = wmma_bf16(aq, bk, c);
#pragma unroll
    for (int j = 0; j < 8; ++j) {
      const int row = j + (half ? 8 : 0);  // C layout: M=j (lanes>=16: +8)
      L[row * LDST + kt * 16 + m] = c[j] * scale;
    }
    if (kt < 16) {  // prefetch adjoin/dist rows (global_prefetch_b8)
      __builtin_prefetch(adj + (qrow_base + kt) * Sn + lane * 32, 0, 0);
      __builtin_prefetch(dist + (qrow_base + kt) * Sn + lane * 32, 0, 0);
    }
  }

  // Phase 2: add biases (coalesced streams), online softmax, emit attn + P
  for (int r = 0; r < 16; ++r) {
    const float* arow = adj + (qrow_base + r) * Sn;
    const float* drow = dist + (qrow_base + r) * Sn;
    if (r + 1 < 16) {
      __builtin_prefetch(arow + Sn + lane * 32, 0, 0);
      __builtin_prefetch(drow + Sn + lane * 32, 0, 0);
    }
    float mloc = -3.0e38f, sloc = 0.0f;
    for (int col = lane; col < Sn; col += 32) {
      float v = L[r * LDST + col] + arow[col] + drow[col];
      L[r * LDST + col] = v;
      const float mnew = fmaxf(mloc, v);
      sloc = sloc * __expf(mloc - mnew) + __expf(v - mnew);
      mloc = mnew;
    }
#pragma unroll
    for (int off = 16; off > 0; off >>= 1) {  // wave32 butterfly reduction
      const float mo = __shfl_xor(mloc, off, 32);
      const float so = __shfl_xor(sloc, off, 32);
      const float mnew = fmaxf(mloc, mo);
      sloc = sloc * __expf(mloc - mnew) + so * __expf(mo - mnew);
      mloc = mnew;
    }
    const float inv = 1.0f / sloc;
    float* orow = attn_out + ((size_t)bh * Sn + q0 + r) * Sn;
    for (int col = lane; col < Sn; col += 32) {
      const float p = __expf(L[r * LDST + col] - mloc) * inv;
      orow[col] = p;              // the dominant 268 MB HBM stream
      L[r * LDST + col] = p;      // kept for P@V
    }
  }

  // Phase 3: ctx = P @ V. A from LDS via float4 (ds_load_b128) + cvt;
  // B from transposed V (key-contiguous per depth column).
  v8f c0 = {}, c1 = {};
  const float* Lrow = L + (size_t)m * LDST;
  for (int kc = 0; kc < Sn; kc += 32) {
    const float* p = Lrow + kc + 8 * half;
    float4 f0 = ((const float4*)p)[0];
    float4 f1 = ((const float4*)p)[1];
    float4 f2 = ((const float4*)(p + 16))[0];
    float4 f3 = ((const float4*)(p + 16))[1];
    v16bf ap;
    ap[0] = (__bf16)f0.x;  ap[1] = (__bf16)f0.y;  ap[2] = (__bf16)f0.z;  ap[3] = (__bf16)f0.w;
    ap[4] = (__bf16)f1.x;  ap[5] = (__bf16)f1.y;  ap[6] = (__bf16)f1.z;  ap[7] = (__bf16)f1.w;
    ap[8] = (__bf16)f2.x;  ap[9] = (__bf16)f2.y;  ap[10] = (__bf16)f2.z; ap[11] = (__bf16)f2.w;
    ap[12] = (__bf16)f3.x; ap[13] = (__bf16)f3.y; ap[14] = (__bf16)f3.z; ap[15] = (__bf16)f3.w;

    const __bf16* vb = vt + (size_t)bh * DEPTHn * Sn + kc;
    v16bf b0 = load_frag_b(vb, Sn, m, half);            // depth cols 0..15
    v16bf b1 = load_frag_b(vb + (size_t)16 * Sn, Sn, m, half);  // 16..31
    c0 = wmma_bf16(ap, b0, c0);
    c1 = wmma_bf16(ap, b1, c1);
  }

  // ctx bf16 in [B,S,D] layout (== transpose(0,2,1,3).reshape)
#pragma unroll
  for (int j = 0; j < 8; ++j) {
    const int q = q0 + j + (half ? 8 : 0);
    const size_t base = ((size_t)b * Sn + q) * Dn + h * DEPTHn;
    ctx[base + m] = (__bf16)c0[j];
    ctx[base + 16 + m] = (__bf16)c1[j];
  }
}

// ---- Output projection: out = ctx @ wo + bo (f32 out) ---------------------
__global__ void out_proj_kernel(const __bf16* __restrict__ ctx,
                                const __bf16* __restrict__ wt,  // [out][in]
                                const float* __restrict__ bias,
                                float* __restrict__ out) {
  const int lane = threadIdx.x & 31;
  const int m = lane & 15, half = lane >> 4;
  const int row0 = blockIdx.x * 16;
  const int col0 = blockIdx.y * 16;

  v8f c = {};
  for (int kc = 0; kc < Dn; kc += 32) {
    v16bf a = load_frag_a(ctx + (size_t)row0 * Dn + kc, Dn, m, half);
    v16bf b = load_frag_b(wt + (size_t)col0 * Dn + kc, Dn, m, half);
    c = wmma_bf16(a, b, c);
  }

  const int col = col0 + m;
  const float bv = bias[col];
#pragma unroll
  for (int j = 0; j < 8; ++j) {
    const int r = row0 + j + (half ? 8 : 0);
    out[(size_t)r * Dn + col] = c[j] + bv;
  }
}

extern "C" void kernel_launch(void* const* d_in, const int* in_sizes, int n_in,
                              void* d_out, int out_size, void* d_ws, size_t ws_size,
                              hipStream_t stream) {
  (void)in_sizes; (void)n_in; (void)out_size; (void)ws_size;
  const float* q    = (const float*)d_in[0];
  const float* k    = (const float*)d_in[1];
  const float* v    = (const float*)d_in[2];
  const float* adjm = (const float*)d_in[3];
  const float* dstm = (const float*)d_in[4];
  const float* wq   = (const float*)d_in[5];
  const float* bq   = (const float*)d_in[6];
  const float* wk   = (const float*)d_in[7];
  const float* bk   = (const float*)d_in[8];
  const float* wv   = (const float*)d_in[9];
  const float* bv   = (const float*)d_in[10];
  const float* wo   = (const float*)d_in[11];
  const float* bo   = (const float*)d_in[12];

  float* out  = (float*)d_out;               // [B,S,D]
  float* attn = out + (size_t)Bn * Sn * Dn;  // [B,H,S,S]

  const size_t qkv_elems = (size_t)Bn * Hn * Sn * DEPTHn;  // 2M bf16
  const size_t w_elems   = (size_t)Dn * Dn;                // 64K bf16
  __bf16* qbf = (__bf16*)d_ws;
  __bf16* kbf = qbf + qkv_elems;
  __bf16* vbf = kbf + qkv_elems;
  __bf16* vtb = vbf + qkv_elems;             // V transposed [B,H,DEPTH,S]
  __bf16* ctx = vtb + qkv_elems;             // [B,S,D]
  __bf16* wqt = ctx + (size_t)Bn * Sn * Dn;
  __bf16* wkt = wqt + w_elems;
  __bf16* wvt = wkt + w_elems;
  __bf16* wot = wvt + w_elems;

  dim3 gW(Dn / 32, Dn / 32);                 // (8,8)
  wtrans_kernel<<<gW, 256, 0, stream>>>(wq, wqt);
  wtrans_kernel<<<gW, 256, 0, stream>>>(wk, wkt);
  wtrans_kernel<<<gW, 256, 0, stream>>>(wv, wvt);
  wtrans_kernel<<<gW, 256, 0, stream>>>(wo, wot);

  dim3 gProj(Bn * Sn / 16, Dn / 16);         // (512, 16) wave tiles
  proj_qkv_kernel<<<gProj, 32, 0, stream>>>(q, wqt, bq, qbf);
  proj_qkv_kernel<<<gProj, 32, 0, stream>>>(k, wkt, bk, kbf);
  proj_qkv_kernel<<<gProj, 32, 0, stream>>>(v, wvt, bv, vbf);

  vtrans_kernel<<<dim3(Sn / 32, Bn * Hn), 256, 0, stream>>>(vbf, vtb);

  const size_t shmem = (size_t)4 * 16 * LDST * sizeof(float);  // 257 KB/WGP
  attn_kernel<<<dim3(Sn / 64, Bn * Hn), 128, shmem, stream>>>(
      qbf, kbf, vtb, adjm, dstm, attn, ctx);

  out_proj_kernel<<<gProj, 32, 0, stream>>>(ctx, wot, bo, out);
}
